// FastCausalSelfAttention_57217554317589
// MI455X (gfx1250) — matrix operand order repaired and checked by reading, not compiled, not run on hardware
//
#include <hip/hip_runtime.h>

// ---------------------------------------------------------------------------
// Fused causal self-attention with RoPE for MI455X (gfx1250, wave32, WMMA).
// Pipeline: f32->bf16 convert, 3x WMMA GEMM (QKV proj) with async-to-LDS
// double-buffered B tiles, RoPE, flash-attention with WMMA (QK^T and P*V)
// and TDM (tensor_load_to_lds) K staging, WMMA GEMM output projection.
// ---------------------------------------------------------------------------

typedef __attribute__((ext_vector_type(16))) __bf16        v16bf;
typedef __attribute__((ext_vector_type(8)))  float         v8f;
typedef __attribute__((ext_vector_type(4)))  unsigned int  v4u;
typedef __attribute__((ext_vector_type(2)))  unsigned int  v2u;
typedef __attribute__((ext_vector_type(8)))  int           v8i;
typedef __attribute__((ext_vector_type(4)))  int           v4i;

union FragB {            // one WMMA 16-bit operand fragment (8 VGPRs / lane)
    v16bf          v;
    unsigned short u[16];
    v4u            q[2]; // two 16-byte chunks
};

#define DEVFN static __device__ __forceinline__

constexpr int Bsz = 4;
constexpr int T   = 2048;
constexpr int D   = 2048;
constexpr int H   = 16;
constexpr int HD  = 128;           // head dim
constexpr int Mrows = Bsz * T;     // 8192 rows in all GEMMs

DEVFN unsigned short f2bf(float f) {
    unsigned u = __builtin_bit_cast(unsigned, f);
    return (unsigned short)((u + 0x7FFFu + ((u >> 16) & 1u)) >> 16);  // RNE
}
DEVFN float bf2f(unsigned short h) {
    unsigned u = ((unsigned)h) << 16;
    return __builtin_bit_cast(float, u);
}
DEVFN v8f vzero8() {
    v8f z;
#pragma unroll
    for (int e = 0; e < 8; ++e) z[e] = 0.0f;
    return z;
}
DEVFN v8f wmma_bf16(v16bf a, v16bf b, v8f c) {
    // D = A(16x32 bf16) * B(32x16 bf16) + C(16x16 f32)
    return __builtin_amdgcn_wmma_f32_16x16x32_bf16(
        false, a, false, b, (short)0, c, false, false);
}

// --- gfx1250 async global->LDS copy (ASYNCcnt) ------------------------------
// LDS byte address = low 32 bits of the flat shared-aperture address.
DEVFN unsigned lds_addr_of(const void* p) {
    return (unsigned)(uintptr_t)p;
}
DEVFN void async_load_b128(unsigned lds_off, const void* gptr) {
    unsigned long long ga = (unsigned long long)(uintptr_t)gptr;
    asm volatile("global_load_async_to_lds_b128 %0, %1, off"
                 :: "v"(lds_off), "v"(ga)
                 : "memory");
}
#if __has_builtin(__builtin_amdgcn_s_wait_asynccnt)
#define WAIT_ASYNC(n) __builtin_amdgcn_s_wait_asynccnt(n)
#else
#define WAIT_ASYNC(n) asm volatile("s_wait_asynccnt %0" :: "i"(n) : "memory")
#endif

// --- gfx1250 Tensor Data Mover (TENSORcnt) ----------------------------------
#if __has_builtin(__builtin_amdgcn_tensor_load_to_lds)
#define HAVE_TDM 1
#else
#define HAVE_TDM 0
#endif

#if __has_builtin(__builtin_amdgcn_s_wait_tensorcnt)
#define WAIT_TENSOR(n) __builtin_amdgcn_s_wait_tensorcnt(n)
#else
#define WAIT_TENSOR(n) asm volatile("s_wait_tensorcnt %0" :: "i"(n) : "memory")
#endif

#if HAVE_TDM
// 2D tile DMA: `rows` rows of `rowElems` bf16 elements, row stride
// `strideElems`, packed contiguously into LDS at lds_addr. One issue per wave
// (TDM ignores EXEC) -> call from a single wave per block.
// This toolchain exposes the 6-arg builtin: (g0 v4u, g1 v8i, g2 v4i, g3 v4i,
// g4 v8i, cpol i32); our 2D descriptor only needs groups 0 and 1.
DEVFN void tdm_load_2d(unsigned lds_addr, const void* gaddr,
                       unsigned rowElems, unsigned rows, unsigned strideElems) {
    unsigned long long ga = (unsigned long long)(uintptr_t)gaddr;
    v4u g0;
    g0[0] = 1u;                                              // count=1 (valid)
    g0[1] = lds_addr;                                        // lds_addr [63:32]
    g0[2] = (unsigned)(ga & 0xFFFFFFFFu);                    // global_addr lo
    g0[3] = (unsigned)((ga >> 32) & 0x01FFFFFFu)             // global_addr hi
          | 0x80000000u;                                     // type=2 ("image")
    v8i g1;
    g1[0] = (int)(1u << 16);            // data_size=1 (2-byte elements)
    g1[1] = (int)(rowElems << 16);      // tensor_dim0[15:0]  @ bits[63:48]
    g1[2] = (int)(rows << 16);          // tensor_dim1[15:0]  @ bits[95:80]
    g1[3] = (int)(rowElems << 16);      // tile_dim0          @ bits[127:112]
    g1[4] = (int)rows;                  // tile_dim1 (tile_dim2 = 0)
    g1[5] = (int)strideElems;           // tensor_dim0_stride[31:0]
    g1[6] = 0;                          // stride0 hi | tensor_dim1_stride lo
    g1[7] = 0;
    v4i g2 = {0, 0, 0, 0};
    v4i g3 = {0, 0, 0, 0};
    v8i g4;
#pragma unroll
    for (int e = 0; e < 8; ++e) g4[e] = 0;
    __builtin_amdgcn_tensor_load_to_lds(g0, g1, g2, g3, g4, 0);
}
#endif

// ---------------------------------------------------------------------------
// f32 -> bf16 bulk convert (vectorized x4)
// ---------------------------------------------------------------------------
__global__ __launch_bounds__(256) void cvt4_kernel(const float* __restrict__ in,
                                                   unsigned short* __restrict__ out,
                                                   int n4) {
    int i = blockIdx.x * 256 + threadIdx.x;
    if (i >= n4) return;
    const float* p = in + (size_t)i * 4;
    float f0 = p[0], f1 = p[1], f2 = p[2], f3 = p[3];
    v2u o;
    o.x = (unsigned)f2bf(f0) | ((unsigned)f2bf(f1) << 16);
    o.y = (unsigned)f2bf(f2) | ((unsigned)f2bf(f3) << 16);
    *(v2u*)(out + (size_t)i * 4) = o;
}

// ---------------------------------------------------------------------------
// WMMA GEMM: C[M,N] = A[M,K](bf16) * W[N,K](bf16)^T,  K = D = 2048
// Block = 128 threads (4 waves); wave w -> 32 rows x 64 cols (8 accumulators,
// 8 WMMAs per K-step). Shared B tile (64x32 bf16, 4KB) staged per block into
// LDS via async global->LDS loads, double-buffered to overlap DMA with WMMA.
// MODE 0: store bf16 into [B,H,T,hd] permuted layout (QKV projections)
// MODE 1: store f32 row-major [M,N] (output projection -> d_out)
// ---------------------------------------------------------------------------
template <int MODE>
__global__ __launch_bounds__(128) void gemm_bf16w(const unsigned short* __restrict__ A,
                                                  const unsigned short* __restrict__ W,
                                                  void* __restrict__ Out) {
    __shared__ unsigned short Bt[2][64 * 32];   // [buf][n][k] 4KB each

    const int tid  = threadIdx.x;
    const int wv   = tid >> 5;        // wave in block
    const int lane = tid & 31;
    const int l16  = lane & 15;
    const int lh   = lane >> 4;       // lane half (0/1)
    const int n0   = blockIdx.x * 64;
    const int m0   = blockIdx.y * 128 + wv * 32;

    const unsigned short* arow0 = A + (size_t)(m0 + l16) * D;
    const unsigned short* arow1 = A + (size_t)(m0 + 16 + l16) * D;

    // B staging: thread copies 32B of row (n0 + tid/2), k-half (tid&1)
    const int br  = tid >> 1;
    const int bh2 = tid & 1;
    const unsigned short* wsrc = W + (size_t)(n0 + br) * D + 16 * bh2;
    const unsigned ldsdst[2] = {
        lds_addr_of(&Bt[0][br * 32 + 16 * bh2]),
        lds_addr_of(&Bt[1][br * 32 + 16 * bh2])
    };

    v8f acc[2][4];
#pragma unroll
    for (int r = 0; r < 2; ++r)
#pragma unroll
        for (int s = 0; s < 4; ++s) acc[r][s] = vzero8();

    // prologue: async-stage tile kt=0 into buffer 0 (2 async ops per wave)
    async_load_b128(ldsdst[0], wsrc);
    async_load_b128(ldsdst[0] + 16, wsrc + 8);

    for (int kt = 0; kt < D; kt += 32) {
        const int cur = (kt >> 5) & 1;
        // stage next tile while computing this one
        if (kt + 32 < D) {
            const int nxt = cur ^ 1;
            async_load_b128(ldsdst[nxt], wsrc + kt + 32);
            async_load_b128(ldsdst[nxt] + 16, wsrc + kt + 40);
            WAIT_ASYNC(2);      // in-order: tile `cur` (2 older ops) has landed
        } else {
            WAIT_ASYNC(0);
        }
        __syncthreads();        // tile `cur` visible to all waves

        // A fragments (16-bit A layout: half0 K{+0..7,+16..23}, half1 K{+8..15,+24..31})
        FragB a0, a1;
        a0.q[0] = *(const v4u*)(arow0 + kt + 8 * lh);
        a0.q[1] = *(const v4u*)(arow0 + kt + 16 + 8 * lh);
        a1.q[0] = *(const v4u*)(arow1 + kt + 8 * lh);
        a1.q[1] = *(const v4u*)(arow1 + kt + 16 + 8 * lh);
        if (kt + 32 < D) {
            __builtin_prefetch(arow0 + kt + 32, 0, 1);   // global_prefetch_b8
            __builtin_prefetch(arow1 + kt + 32, 0, 1);
        }
#pragma unroll
        for (int s = 0; s < 4; ++s) {
            const unsigned short* bp = &Bt[cur][(s * 16 + l16) * 32];
            FragB b;
            b.q[0] = *(const v4u*)(bp + 8 * lh);
            b.q[1] = *(const v4u*)(bp + 16 + 8 * lh);
            acc[0][s] = wmma_bf16(a0.v, b.v, acc[0][s]);
            acc[1][s] = wmma_bf16(a1.v, b.v, acc[1][s]);
        }
        __syncthreads();        // all done reading `cur` before it is re-staged
    }

    if (MODE == 0) {
        unsigned short* out = (unsigned short*)Out;
#pragma unroll
        for (int s = 0; s < 4; ++s) {
            int n  = n0 + s * 16 + l16;
            int h  = n >> 7;       // / HD
            int dd = n & 127;      // % HD
#pragma unroll
            for (int r = 0; r < 2; ++r) {
#pragma unroll
                for (int i = 0; i < 8; ++i) {
                    int m  = m0 + r * 16 + i + 8 * lh;  // C layout rows
                    int bb = m >> 11;                    // / T
                    int t  = m & 2047;                   // % T
                    out[(((size_t)(bb * H + h)) * T + t) * HD + dd] =
                        f2bf(acc[r][s][i]);
                }
            }
        }
    } else {
        float* out = (float*)Out;
#pragma unroll
        for (int s = 0; s < 4; ++s) {
#pragma unroll
            for (int r = 0; r < 2; ++r) {
#pragma unroll
                for (int i = 0; i < 8; ++i) {
                    out[(size_t)(m0 + r * 16 + i + 8 * lh) * D + n0 + s * 16 + l16] =
                        acc[r][s][i];
                }
            }
        }
    }
}

// ---------------------------------------------------------------------------
// RoPE on Q and K (bf16 in-place), layout [B,H,T,hd], pairs (d, d+64)
// ---------------------------------------------------------------------------
__global__ __launch_bounds__(256) void rope_kernel(unsigned short* __restrict__ q,
                                                   unsigned short* __restrict__ k) {
    int idx = blockIdx.x * 256 + threadIdx.x;       // over B*H*T*(HD/2)
    int d   = idx & 63;
    int t   = (idx >> 6) & (T - 1);
    int bh  = idx >> 17;
    size_t base = ((size_t)bh * T + t) * HD;

    float freq = __powf(10000.0f, -(float)d / 64.0f);   // base^(-2d/hd)
    float ang  = (float)t * freq;
    float s, c;
    __sincosf(ang, &s, &c);

    float q1 = bf2f(q[base + d]), q2 = bf2f(q[base + d + 64]);
    q[base + d]      = f2bf(q1 * c - q2 * s);
    q[base + d + 64] = f2bf(q2 * c + q1 * s);

    float k1 = bf2f(k[base + d]), k2 = bf2f(k[base + d + 64]);
    k[base + d]      = f2bf(k1 * c - k2 * s);
    k[base + d + 64] = f2bf(k2 * c + k1 * s);
}

// ---------------------------------------------------------------------------
// Flash-style causal attention, WMMA for QK^T and P*V.
// grid = (T/64, B*H), block = 128 (4 waves). Wave w owns q rows [q0, q0+16).
// K tile staged row-major via the Tensor Data Mover (one descriptor per
// block, issued by wave 0, s_wait_tensorcnt) with async-to-LDS fallback;
// V tile transposed through VGPRs. Both shared by the 4 waves.
// ---------------------------------------------------------------------------
__global__ __launch_bounds__(128) void attn_kernel(const unsigned short* __restrict__ q,
                                                   const unsigned short* __restrict__ k,
                                                   const unsigned short* __restrict__ v,
                                                   unsigned short* __restrict__ y) {
    __shared__ unsigned short Kt[32 * 128];     // [key][dim]
    __shared__ unsigned short Vt[128 * 32];     // [dim][key] (transposed)
    __shared__ unsigned short Pt[4][16 * 32];   // per-wave P staging (C->A relayout)

    const int tid  = threadIdx.x;
    const int wv   = tid >> 5;
    const int lane = tid & 31;
    const int l16  = lane & 15;
    const int lh   = lane >> 4;
    const int bh   = blockIdx.y;
    const size_t bhOff = (size_t)bh * T * HD;
    const int qBase = blockIdx.x * 64;
    const int myQ   = qBase + wv * 16;

    // Q fragments: 4 x (16x32) A-layout tiles covering hd = 128
    v16bf aq[4];
    {
        const unsigned short* qrow = q + bhOff + (size_t)(myQ + l16) * HD;
#pragma unroll
        for (int f = 0; f < 4; ++f) {
            FragB a;
            a.q[0] = *(const v4u*)(qrow + f * 32 + 8 * lh);
            a.q[1] = *(const v4u*)(qrow + f * 32 + 16 + 8 * lh);
            aq[f] = a.v;
        }
    }

    float mrow[8], lrow[8];
    v8f o[8];
#pragma unroll
    for (int i = 0; i < 8; ++i) { mrow[i] = -1e30f; lrow[i] = 0.0f; }
#pragma unroll
    for (int s = 0; s < 8; ++s) o[s] = vzero8();

    const float scale = 0.08838834764831845f;   // 1/sqrt(128)
    const int kEnd = qBase + 64;                // causal bound for the block

    for (int j0 = 0; j0 < kEnd; j0 += 32) {
        __syncthreads();
        // --- stage K rows j0..j0+31 (row-major, 8KB) ---
#if HAVE_TDM
        if (wv == 0) {
            // one 2D descriptor: 32 rows x 128 bf16, stride HD, packed to LDS
            tdm_load_2d(lds_addr_of(Kt),
                        k + bhOff + (size_t)j0 * HD, HD, 32, HD);
        }
#else
#pragma unroll
        for (int it = 0; it < 4; ++it) {
            int e = (it * 128 + tid) * 8;       // element index in 32x128 tile
            int r = e >> 7, c = e & 127;
            async_load_b128(lds_addr_of(Kt + e),
                            k + bhOff + (size_t)(j0 + r) * HD + c);
        }
#endif
        // --- cooperative stage: V transposed [dim][key] (overlaps K DMA) ---
#pragma unroll
        for (int it = 0; it < 8; ++it) {
            int e = (it * 128 + tid) * 4;
            int r = e >> 7, c = e & 127;
            v2u d = *(const v2u*)(v + bhOff + (size_t)(j0 + r) * HD + c);
            Vt[(c + 0) * 32 + r] = (unsigned short)(d.x & 0xFFFFu);
            Vt[(c + 1) * 32 + r] = (unsigned short)(d.x >> 16);
            Vt[(c + 2) * 32 + r] = (unsigned short)(d.y & 0xFFFFu);
            Vt[(c + 3) * 32 + r] = (unsigned short)(d.y >> 16);
        }
#if HAVE_TDM
        if (wv == 0) { WAIT_TENSOR(0); }
#else
        WAIT_ASYNC(0);
#endif
        __syncthreads();

        if (j0 >= myQ + 16) continue;           // beyond this wave's diagonal

        // --- S = Q * K^T for keys [j0, j0+32): two 16x16 C tiles ---
        v8f sa[2];
#pragma unroll
        for (int st = 0; st < 2; ++st) {
            v8f s_acc = vzero8();
#pragma unroll
            for (int f = 0; f < 4; ++f) {
                const unsigned short* kp = Kt + (st * 16 + l16) * 128 + f * 32;
                FragB b;
                b.q[0] = *(const v4u*)(kp + 8 * lh);
                b.q[1] = *(const v4u*)(kp + 16 + 8 * lh);
                s_acc = wmma_bf16(aq[f], b.v, s_acc);
            }
            sa[st] = s_acc;
        }

        // --- online softmax (rows split per VGPR and per 16-lane half) ---
#pragma unroll
        for (int i = 0; i < 8; ++i) {
            int qrowIdx = myQ + i + 8 * lh;
            float v0 = sa[0][i] * scale;
            float v1 = sa[1][i] * scale;
            if (j0 + l16 > qrowIdx)      v0 = -1e30f;   // causal mask
            if (j0 + 16 + l16 > qrowIdx) v1 = -1e30f;
            float mx = fmaxf(v0, v1);
            mx = fmaxf(mx, __shfl_xor(mx, 1, 32));
            mx = fmaxf(mx, __shfl_xor(mx, 2, 32));
            mx = fmaxf(mx, __shfl_xor(mx, 4, 32));
            mx = fmaxf(mx, __shfl_xor(mx, 8, 32));
            float mnew  = fmaxf(mrow[i], mx);
            float alpha = __expf(mrow[i] - mnew);
            float p0 = __expf(v0 - mnew);
            float p1 = __expf(v1 - mnew);
            float rs = p0 + p1;
            rs += __shfl_xor(rs, 1, 32);
            rs += __shfl_xor(rs, 2, 32);
            rs += __shfl_xor(rs, 4, 32);
            rs += __shfl_xor(rs, 8, 32);
            lrow[i] = lrow[i] * alpha + rs;
            mrow[i] = mnew;
#pragma unroll
            for (int s = 0; s < 8; ++s) o[s][i] *= alpha;
            // stage P (C layout -> LDS row-major 16x32)
            Pt[wv][(i + 8 * lh) * 32 + l16]      = f2bf(p0);
            Pt[wv][(i + 8 * lh) * 32 + 16 + l16] = f2bf(p1);
        }

        // --- reload P as an A fragment (16x32) ---
        FragB pa;
        pa.q[0] = *(const v4u*)(&Pt[wv][l16 * 32 + 8 * lh]);
        pa.q[1] = *(const v4u*)(&Pt[wv][l16 * 32 + 16 + 8 * lh]);

        // --- O += P * V  (8 x 16-col tiles over hd=128) ---
#pragma unroll
        for (int s = 0; s < 8; ++s) {
            const unsigned short* vp = Vt + (s * 16 + l16) * 32;
            FragB b;
            b.q[0] = *(const v4u*)(vp + 8 * lh);
            b.q[1] = *(const v4u*)(vp + 16 + 8 * lh);
            o[s] = wmma_bf16(pa.v, b.v, o[s]);
        }
    }

    // --- epilogue: normalize and store y[B,T,D] (bf16) ---
    const int bb = bh / H, hh = bh % H;
#pragma unroll
    for (int s = 0; s < 8; ++s) {
#pragma unroll
        for (int i = 0; i < 8; ++i) {
            int qrowIdx = myQ + i + 8 * lh;
            float val = o[s][i] / lrow[i];
            y[((size_t)(bb * T + qrowIdx)) * D + hh * HD + s * 16 + l16] = f2bf(val);
        }
    }
}

// ---------------------------------------------------------------------------
// Launcher
// ---------------------------------------------------------------------------
extern "C" void kernel_launch(void* const* d_in, const int* in_sizes, int n_in,
                              void* d_out, int out_size, void* d_ws, size_t ws_size,
                              hipStream_t stream) {
    const float* x  = (const float*)d_in[0];
    const float* wq = (const float*)d_in[1];
    const float* wk = (const float*)d_in[2];
    const float* wv = (const float*)d_in[3];
    const float* wo = (const float*)d_in[4];
    float* out = (float*)d_out;

    const size_t MD = (size_t)Mrows * D;   // 16,777,216 elements
    const size_t DD = (size_t)D * D;       //  4,194,304 elements

    unsigned short* ws  = (unsigned short*)d_ws;
    unsigned short* xbf = ws;              // [M,D]  bf16 activations
    unsigned short* wqb = xbf + MD;        // [D,D]  bf16 weights
    unsigned short* wkb = wqb + DD;
    unsigned short* wvb = wkb + DD;
    unsigned short* wob = wvb + DD;
    unsigned short* qb  = wob + DD;        // [B,H,T,hd]
    unsigned short* kb  = qb + MD;
    unsigned short* vb  = kb + MD;
    unsigned short* yb  = vb + MD;         // [B,T,D] attention output

    // 1) f32 -> bf16 converts
    cvt4_kernel<<<(int)(MD / 4 / 256), 256, 0, stream>>>(x,  xbf, (int)(MD / 4));
    cvt4_kernel<<<(int)(DD / 4 / 256), 256, 0, stream>>>(wq, wqb, (int)(DD / 4));
    cvt4_kernel<<<(int)(DD / 4 / 256), 256, 0, stream>>>(wk, wkb, (int)(DD / 4));
    cvt4_kernel<<<(int)(DD / 4 / 256), 256, 0, stream>>>(wv, wvb, (int)(DD / 4));
    cvt4_kernel<<<(int)(DD / 4 / 256), 256, 0, stream>>>(wo, wob, (int)(DD / 4));

    // 2) QKV projections (WMMA), output in [B,H,T,hd] bf16
    dim3 gg(D / 64, Mrows / 128);
    gemm_bf16w<0><<<gg, 128, 0, stream>>>(xbf, wqb, (void*)qb);
    gemm_bf16w<0><<<gg, 128, 0, stream>>>(xbf, wkb, (void*)kb);
    gemm_bf16w<0><<<gg, 128, 0, stream>>>(xbf, wvb, (void*)vb);

    // 3) RoPE on Q,K
    rope_kernel<<<(Bsz * H * T * (HD / 2)) / 256, 256, 0, stream>>>(qb, kb);

    // 4) Causal attention (WMMA flash, TDM K staging)
    attn_kernel<<<dim3(T / 64, Bsz * H), 128, 0, stream>>>(qb, kb, vb, yb);

    // 5) Output projection (WMMA), f32 store to d_out
    gemm_bf16w<1><<<gg, 128, 0, stream>>>(yb, wob, (void*)out);
}
